// UniModalMamba_89404039233694
// MI455X (gfx1250) — compile-verified, matrix operand order checked by reading, not run on hardware
//
#include <hip/hip_runtime.h>
#include <hip/hip_bf16.h>
#include <cstdint>

// ---- model dims ----
#define BSZ 4
#define D_ 768
#define DEPTH_ 8
#define DINNER 1536
#define DTRANK 48
#define DSTATE 16
#define DCONV 4
#define LSEQ 288              // 16 frames * 18 tokens
#define NTOK (BSZ * LSEQ)     // 1152
#define XLD 128               // padded x_proj output width (dt|B|C|zeros)
#define DTK 64                // padded dt_proj K
#define SWPAD 40              // LDS W-panel row stride in halves (bank de-stagger)

typedef __attribute__((ext_vector_type(16))) _Float16 v16h;
typedef __attribute__((ext_vector_type(8)))  _Float16 v8h;
typedef __attribute__((ext_vector_type(8)))  float    v8f;
typedef int v4i_gcc __attribute__((vector_size(16)));   // matches builtin param type

#if __has_builtin(__builtin_amdgcn_global_load_async_to_lds_b128) && \
    __has_builtin(__builtin_amdgcn_s_wait_asynccnt)
#define ASYNC_LDS 1
#else
#define ASYNC_LDS 0
#endif

// ============================================================
// Guard-free GEMM on pre-converted f16 operands, f32 accumulate:
//   C[M,N] = A[M,K] * W[N,K]^T
// REQUIRES: M%128==0, K%32==0, N%64==0, lda/ldw %8==0, 16B bases.
// Block = 8 waves stacked along M sharing ONE 64-col W strip:
// the 64x32 W panel is staged in LDS (async global->LDS copy,
// double buffered) once per block, cutting W L2 traffic 8x.
// Fragment layout per CDNA5 ISA 7.12.2: lane (l&15)=row,
// lane>>4 selects K half (+8); VGPR 0-3 = K khalf+0..7 (contig),
// VGPR 4-7 = K khalf+16..23 (contig) -> two 16B loads per fragment.
// ============================================================
__global__ __launch_bounds__(256) void gemm_wmma_f16(
    const _Float16* __restrict__ A, const _Float16* __restrict__ W,
    float* __restrict__ C, int N, int K, int lda, int ldw, int ldc)
{
    __shared__ alignas(16) _Float16 sw[2][64 * SWPAD];

    const int nBlocks = N >> 6;
    const int bm = blockIdx.x / nBlocks;
    const int nb = blockIdx.x - bm * nBlocks;
    const int wv   = threadIdx.x >> 5;
    const int lane = threadIdx.x & 31;
    const int m0 = (bm * 8 + wv) * 16;
    const int n0 = nb * 64;

    // ---- staging assignment: 256 threads x 16B cover 64 rows x 32 halves
    const int srow   = threadIdx.x >> 2;        // 0..63
    const int schunk = (threadIdx.x & 3) << 3;  // 0,8,16,24 halves
    const _Float16* Wsrc = W + (size_t)(n0 + srow) * ldw + schunk;
    _Float16* sdst0 = &sw[0][srow * SWPAD + schunk];
    _Float16* sdst1 = &sw[1][srow * SWPAD + schunk];

#if ASYNC_LDS
    // AS1 pointer: global AS is identity-mapped inside generic -> inttoptr ok.
    // AS3 pointer: ISA 10.2, flat-LDS aperture uses addr[31:0] as the LDS
    // address, so the low 32 bits of the generic pointer ARE the LDS address.
#define STAGE(buf, koff)                                                      \
    __builtin_amdgcn_global_load_async_to_lds_b128(                           \
        (__attribute__((address_space(1))) v4i_gcc*)(unsigned long long)      \
            (uintptr_t)(Wsrc + (koff)),                                       \
        (__attribute__((address_space(3))) v4i_gcc*)(unsigned int)            \
            (uintptr_t)((buf) ? sdst1 : sdst0), 0, 0)
#else
#define STAGE(buf, koff)                                                      \
    *(v8h*)((buf) ? sdst1 : sdst0) = *(const v8h*)(Wsrc + (koff))
#endif

    const int khalf = (lane >> 4) << 3;   // 0 or 8
    const _Float16* Arow = A + (size_t)(m0 + (lane & 15)) * lda + khalf;

    v8f acc[4] = {};

    const int P = K >> 5;
    STAGE(0, 0);
    for (int p = 0; p < P; ++p) {
        const int k0 = p << 5;
        if (p + 1 < P) {
            STAGE((p + 1) & 1, k0 + 32);
#if ASYNC_LDS
            __builtin_amdgcn_s_wait_asynccnt(1);  // oldest panel landed
#endif
        } else {
#if ASYNC_LDS
            __builtin_amdgcn_s_wait_asynccnt(0);
#endif
        }
        __syncthreads();                          // panel p visible to all waves

        if (k0 + 32 < K) __builtin_prefetch(Arow + k0 + 32, 0, 1);
        const v8h a_lo = *(const v8h*)(Arow + k0);
        const v8h a_hi = *(const v8h*)(Arow + k0 + 16);
        const v16h af = __builtin_shufflevector(a_lo, a_hi,
            0,1,2,3,4,5,6,7,8,9,10,11,12,13,14,15);

        const _Float16* swc = sw[p & 1];
#pragma unroll
        for (int t = 0; t < 4; ++t) {
            const int rowl = t * 16 + (lane & 15);
            const v8h b_lo = *(const v8h*)(swc + rowl * SWPAD + khalf);
            const v8h b_hi = *(const v8h*)(swc + rowl * SWPAD + khalf + 16);
            const v16h bf = __builtin_shufflevector(b_lo, b_hi,
                0,1,2,3,4,5,6,7,8,9,10,11,12,13,14,15);
            acc[t] = __builtin_amdgcn_wmma_f32_16x16x32_f16(
                false, af, false, bf, (short)0, acc[t], false, false);
        }
        __syncthreads();                          // buffer safe to restage
    }
#undef STAGE

    // C/D layout: VGPR r -> M = r + 8*(lane>>4), N = lane&15
    const int rowBase = m0 + ((lane >> 4) << 3);
    float* Cp = C + (size_t)rowBase * ldc + n0 + (lane & 15);
#pragma unroll
    for (int r = 0; r < 8; ++r) {
        Cp[(size_t)r * ldc     ] = acc[0][r];
        Cp[(size_t)r * ldc + 16] = acc[1][r];
        Cp[(size_t)r * ldc + 32] = acc[2][r];
        Cp[(size_t)r * ldc + 48] = acc[3][r];
    }
}

// ---- f32 -> f16 with 2D zero padding ----
__global__ __launch_bounds__(256) void f32_to_f16_pad_kernel(
    const float* __restrict__ src, _Float16* __restrict__ dst,
    int srows, int scols, int slds, int dcols, int total)
{
    const int idx = blockIdx.x * 256 + threadIdx.x;
    if (idx >= total) return;
    const int r = idx / dcols, c = idx - r * dcols;
    const float v = (r < srows && c < scols) ? src[(size_t)r * slds + c] : 0.f;
    dst[idx] = (_Float16)v;
}

// ============================================================
// Patch embedding + positional embeddings + token assembly.
// ============================================================
__global__ __launch_bounds__(256) void patch_embed_kernel(
    const float* __restrict__ x,       // (4, 64, 1024)
    const float* __restrict__ pw,      // (768, 1, 16, 16)
    const float* __restrict__ pb,
    const float* __restrict__ pos,     // (1, 16, 768)
    const float* __restrict__ tpos,    // (1, 16, 768)
    const float* __restrict__ cls,
    const float* __restrict__ glob,
    float* __restrict__ h0)            // (NTOK, 768)
{
    __shared__ float sx[256];
    const int blk = blockIdx.x;
    const int b = blk / LSEQ, s = blk - b * LSEQ;
    const int f = s / 18, r = s - f * 18;
    const int tid = threadIdx.x;
    float* outrow = h0 + (size_t)blk * D_;

    if (r == 0) { for (int d = tid; d < D_; d += 256) outrow[d] = cls[d];  return; }
    if (r == 17){ for (int d = tid; d < D_; d += 256) outrow[d] = glob[d]; return; }

    const int q = r - 1;
    const int t256 = f * 16 + q;
    const int hy = t256 >> 2, wx = t256 & 3;
    {   // sx[i*16+j] = x[b, wx*16+j, hy*16+i]
        const int i = tid >> 4, j = tid & 15;
        sx[tid] = x[(size_t)b * 65536 + (size_t)(wx * 16 + j) * 1024 + (hy * 16 + i)];
    }
    __syncthreads();
    for (int d = tid; d < D_; d += 256) {
        const float* wr = pw + (size_t)d * 256;
        float acc = pb[d];
#pragma unroll 8
        for (int e = 0; e < 256; ++e) acc += sx[e] * wr[e];
        outrow[d] = acc + pos[q * D_ + d] + tpos[f * D_ + d];
    }
}

// ============================================================
// Residual update + RMSNorm; writes f16 (for GEMM) and/or f32.
// ============================================================
__global__ __launch_bounds__(256) void rmsnorm_kernel(
    const float* __restrict__ hidden, float* __restrict__ resbuf,
    const float* __restrict__ w, _Float16* __restrict__ outh,
    float* __restrict__ outf, int addRes, int writeRes)
{
    __shared__ float red[256];
    const int row = blockIdx.x;
    const int tid = threadIdx.x;
    const float* hr = hidden + (size_t)row * D_;
    float* rr = resbuf + (size_t)row * D_;

    float vals[3];
    float ss = 0.f;
#pragma unroll
    for (int e = 0; e < 3; ++e) {
        const int d = tid + e * 256;
        float v = hr[d];
        if (addRes) v += rr[d];
        vals[e] = v;
        ss += v * v;
    }
    red[tid] = ss;
    __syncthreads();
    for (int off = 128; off > 0; off >>= 1) {
        if (tid < off) red[tid] += red[tid + off];
        __syncthreads();
    }
    const float inv = rsqrtf(red[0] * (1.f / (float)D_) + 1e-5f);
#pragma unroll
    for (int e = 0; e < 3; ++e) {
        const int d = tid + e * 256;
        if (writeRes) rr[d] = vals[e];
        const float o = vals[e] * inv * w[d];
        if (outh) outh[(size_t)row * D_ + d] = (_Float16)o;
        if (outf) outf[(size_t)row * D_ + d] = o;
    }
}

// ============================================================
// Causal depthwise conv (k=4) + SiLU; rev=1 walks the reversed
// sequence (output in scan order). Writes f32 (scan u) + f16 (GEMM A).
// ============================================================
__global__ __launch_bounds__(256) void conv_silu_kernel(
    const float* __restrict__ xz, const float* __restrict__ cw,
    const float* __restrict__ cb, float* __restrict__ xc,
    _Float16* __restrict__ xch, int rev)
{
    const int idx = blockIdx.x * 256 + threadIdx.x;
    if (idx >= NTOK * DINNER) return;
    const int c = idx % DINNER;
    const int row = idx / DINNER;
    const int b = row / LSEQ, t = row - b * LSEQ;

    float s = cb[c];
#pragma unroll
    for (int k = 0; k < DCONV; ++k) {
        const int tp = t - (DCONV - 1) + k;
        if (tp >= 0) {
            const int src = rev ? (LSEQ - 1 - tp) : tp;
            s += cw[c * DCONV + k] * xz[(size_t)(b * LSEQ + src) * (2 * DINNER) + c];
        }
    }
    const float y = s / (1.f + __expf(-s));
    xc[idx] = y;
    xch[idx] = (_Float16)y;
}

// ---- dt = softplus(raw + bias) ----
__global__ __launch_bounds__(256) void bias_softplus_kernel(
    const float* __restrict__ raw, const float* __restrict__ db,
    float* __restrict__ dt)
{
    const int idx = blockIdx.x * 256 + threadIdx.x;
    if (idx >= NTOK * DINNER) return;
    const int c = idx % DINNER;
    const float v = raw[idx] + db[c];
    dt[idx] = (v > 20.f) ? v : log1pf(__expf(v));
}

// ============================================================
// Selective scan: one thread per (batch, channel); 16-state
// recurrence over L=288. B/C live in dbl (stride XLD) cols 48/64.
// ============================================================
__global__ __launch_bounds__(256) void ssm_scan_kernel(
    const float* __restrict__ u, const float* __restrict__ dt,
    const float* __restrict__ dbl, const float* __restrict__ A_log,
    const float* __restrict__ Dp, float* __restrict__ y)
{
    const int idx = blockIdx.x * 256 + threadIdx.x;
    if (idx >= BSZ * DINNER) return;
    const int b = idx / DINNER, c = idx - b * DINNER;

    float Ar[DSTATE];
#pragma unroll
    for (int n = 0; n < DSTATE; ++n) Ar[n] = -__expf(A_log[c * DSTATE + n]);
    const float Dc = Dp[c];
    float h[DSTATE];
#pragma unroll
    for (int n = 0; n < DSTATE; ++n) h[n] = 0.f;

    for (int t = 0; t < LSEQ; ++t) {
        const int row = b * LSEQ + t;
        const float ut  = u [(size_t)row * DINNER + c];
        const float dtt = dt[(size_t)row * DINNER + c];
        const float dtu = dtt * ut;
        const float* bc = dbl + (size_t)row * XLD;
        float acc = 0.f;
#pragma unroll
        for (int n = 0; n < DSTATE; ++n) {
            const float dA = __expf(dtt * Ar[n]);
            h[n] = dA * h[n] + dtu * bc[DTRANK + n];
            acc += h[n] * bc[DTRANK + DSTATE + n];
        }
        y[(size_t)row * DINNER + c] = acc + ut * Dc;
    }
}

// ---- g = (yf + reverse(yb)) * silu(z), written as f16 GEMM operand ----
__global__ __launch_bounds__(256) void combine_kernel(
    const float* __restrict__ yf, const float* __restrict__ yb,
    const float* __restrict__ xz, _Float16* __restrict__ g)
{
    const int idx = blockIdx.x * 256 + threadIdx.x;
    if (idx >= NTOK * DINNER) return;
    const int c = idx % DINNER;
    const int row = idx / DINNER;
    const int b = row / LSEQ, t = row - b * LSEQ;
    const float zf = xz[(size_t)row * (2 * DINNER) + DINNER + c];
    const float sig = 1.f / (1.f + __expf(-zf));
    const float yv = yf[idx] + yb[(size_t)(b * LSEQ + (LSEQ - 1 - t)) * DINNER + c];
    g[idx] = (_Float16)(yv * zf * sig);
}

// ============================================================
static inline int gemm_blocks(int N) {
    return (NTOK / 16 / 8) * (N / 64);   // 9 * nBlocks, exact
}
static inline int ew_blocks(int n) { return (n + 255) / 256; }

extern "C" void kernel_launch(void* const* d_in, const int* in_sizes, int n_in,
                              void* d_out, int out_size, void* d_ws, size_t ws_size,
                              hipStream_t stream) {
    (void)in_sizes; (void)n_in; (void)out_size; (void)ws_size;
    const float* x          = (const float*)d_in[0];
    const float* patch_w    = (const float*)d_in[1];
    const float* patch_b    = (const float*)d_in[2];
    const float* pos_embed  = (const float*)d_in[3];
    const float* tpos       = (const float*)d_in[4];
    const float* cls_tok    = (const float*)d_in[5];
    const float* glob_tok   = (const float*)d_in[6];
    const float* norm_w     = (const float*)d_in[7];
    const float* in_proj_w  = (const float*)d_in[8];
    const float* conv_w     = (const float*)d_in[9];
    const float* conv_b     = (const float*)d_in[10];
    const float* xproj_w    = (const float*)d_in[11];
    const float* dtproj_w   = (const float*)d_in[12];
    const float* dtproj_b   = (const float*)d_in[13];
    const float* A_log      = (const float*)d_in[14];
    const float* Dp         = (const float*)d_in[15];
    const float* conv_w_b   = (const float*)d_in[16];
    const float* conv_b_b   = (const float*)d_in[17];
    const float* xproj_w_b  = (const float*)d_in[18];
    const float* dtproj_w_b = (const float*)d_in[19];
    const float* dtproj_b_b = (const float*)d_in[20];
    const float* A_log_b    = (const float*)d_in[21];
    const float* Dp_b       = (const float*)d_in[22];
    const float* out_proj_w = (const float*)d_in[23];
    const float* normf_w    = (const float*)d_in[24];

    float* ws = (float*)d_ws;
    size_t off = 0;   // float units; keep 32B alignment (mult of 8 floats)
    auto allocF = [&](size_t n) { float* p = ws + off; off += (n + 7) & ~(size_t)7; return p; };
    auto allocH = [&](size_t n) { return (_Float16*)allocF((n + 1) / 2); };

    // f32 buffers
    float* hidden = allocF((size_t)NTOK * D_);
    float* resbuf = allocF((size_t)NTOK * D_);
    float* xz     = allocF((size_t)NTOK * 2 * DINNER);
    float* xc     = allocF((size_t)NTOK * DINNER);
    float* dbl    = allocF((size_t)NTOK * XLD);
    float* dtraw  = allocF((size_t)NTOK * DINNER);
    float* dt     = allocF((size_t)NTOK * DINNER);
    float* yf     = allocF((size_t)NTOK * DINNER);
    float* yb     = allocF((size_t)NTOK * DINNER);
    // f16 buffers (GEMM operands)
    _Float16* hn_h   = allocH((size_t)NTOK * D_);
    _Float16* xc_h   = allocH((size_t)NTOK * DINNER);
    _Float16* g_h    = allocH((size_t)NTOK * DINNER);
    _Float16* dtin_h = allocH((size_t)NTOK * DTK);
    _Float16* w_in_h = allocH((size_t)2 * DINNER * D_);
    _Float16* w_xp_h = allocH((size_t)XLD * DINNER);
    _Float16* w_dt_h = allocH((size_t)DINNER * DTK);
    _Float16* w_op_h = allocH((size_t)D_ * DINNER);

    const int EW   = ew_blocks(NTOK * DINNER);   // 6912
    const int SCAN = ew_blocks(BSZ * DINNER);    // 24

    patch_embed_kernel<<<NTOK, 256, 0, stream>>>(
        x, patch_w, patch_b, pos_embed, tpos, cls_tok, glob_tok, hidden);

    for (int i = 0; i < DEPTH_; ++i) {
        rmsnorm_kernel<<<NTOK, 256, 0, stream>>>(
            hidden, resbuf, norm_w + i * D_, hn_h, nullptr, i > 0, 1);

        // in_proj: (1152x3072) = hn(1152x768) * W(3072x768)^T
        f32_to_f16_pad_kernel<<<ew_blocks(2 * DINNER * D_), 256, 0, stream>>>(
            in_proj_w + (size_t)i * 2 * DINNER * D_, w_in_h,
            2 * DINNER, D_, D_, D_, 2 * DINNER * D_);
        gemm_wmma_f16<<<gemm_blocks(2 * DINNER), 256, 0, stream>>>(
            hn_h, w_in_h, xz, 2 * DINNER, D_, D_, D_, 2 * DINNER);

        for (int dir = 0; dir < 2; ++dir) {
            const float* cw  = dir ? conv_w_b   : conv_w;
            const float* cb  = dir ? conv_b_b   : conv_b;
            const float* xpw = dir ? xproj_w_b  : xproj_w;
            const float* dtw = dir ? dtproj_w_b : dtproj_w;
            const float* dtb = dir ? dtproj_b_b : dtproj_b;
            const float* Al  = dir ? A_log_b    : A_log;
            const float* Dpp = dir ? Dp_b       : Dp;
            float* yo        = dir ? yb : yf;

            conv_silu_kernel<<<EW, 256, 0, stream>>>(
                xz, cw + (size_t)i * DINNER * DCONV, cb + (size_t)i * DINNER,
                xc, xc_h, dir);

            // x_proj: W padded 80 -> 128 zero rows; dbl stride 128
            f32_to_f16_pad_kernel<<<ew_blocks(XLD * DINNER), 256, 0, stream>>>(
                xpw + (size_t)i * (DTRANK + 2 * DSTATE) * DINNER, w_xp_h,
                DTRANK + 2 * DSTATE, DINNER, DINNER, DINNER, XLD * DINNER);
            gemm_wmma_f16<<<gemm_blocks(XLD), 256, 0, stream>>>(
                xc_h, w_xp_h, dbl, XLD, DINNER, DINNER, DINNER, XLD);

            // dt_proj: K padded 48 -> 64 on both operands
            f32_to_f16_pad_kernel<<<ew_blocks(NTOK * DTK), 256, 0, stream>>>(
                dbl, dtin_h, NTOK, DTRANK, XLD, DTK, NTOK * DTK);
            f32_to_f16_pad_kernel<<<ew_blocks(DINNER * DTK), 256, 0, stream>>>(
                dtw + (size_t)i * DINNER * DTRANK, w_dt_h,
                DINNER, DTRANK, DTRANK, DTK, DINNER * DTK);
            gemm_wmma_f16<<<gemm_blocks(DINNER), 256, 0, stream>>>(
                dtin_h, w_dt_h, dtraw, DINNER, DTK, DTK, DTK, DINNER);

            bias_softplus_kernel<<<EW, 256, 0, stream>>>(
                dtraw, dtb + (size_t)i * DINNER, dt);
            ssm_scan_kernel<<<SCAN, 256, 0, stream>>>(
                xc, dt, dbl, Al + (size_t)i * DINNER * DSTATE,
                Dpp + (size_t)i * DINNER, yo);
        }

        // gate + out_proj
        combine_kernel<<<EW, 256, 0, stream>>>(yf, yb, xz, g_h);
        f32_to_f16_pad_kernel<<<ew_blocks(D_ * DINNER), 256, 0, stream>>>(
            out_proj_w + (size_t)i * D_ * DINNER, w_op_h,
            D_, DINNER, DINNER, DINNER, D_ * DINNER);
        gemm_wmma_f16<<<gemm_blocks(D_), 256, 0, stream>>>(
            g_h, w_op_h, hidden, D_, DINNER, DINNER, DINNER, D_);
    }

    rmsnorm_kernel<<<NTOK, 256, 0, stream>>>(
        hidden, resbuf, normf_w, nullptr, (float*)d_out, 1, 0);
}
